// ExpertChoiceMoRBlock_81784767251164
// MI455X (gfx1250) — matrix operand order repaired and checked by reading, not compiled, not run on hardware
//
#include <hip/hip_runtime.h>
#include <math.h>

// ---------------- problem constants ----------------
#define BB    4
#define SEQ   4096
#define DIM   2048
#define FF    8192
#define KTOP  2048          // CAPACITY_FACTOR * SEQ
#define ALPHA 0.1f

// ---------------- WMMA types ----------------
typedef __attribute__((ext_vector_type(16))) __bf16 bf16x16;
typedef __attribute__((ext_vector_type(8)))  __bf16 bf16x8;
typedef __attribute__((ext_vector_type(8)))  float  f32x8;

__device__ __forceinline__ bf16x16 cat8(bf16x8 lo, bf16x8 hi) {
    return __builtin_shufflevector(lo, hi, 0,1,2,3,4,5,6,7,8,9,10,11,12,13,14,15);
}

// fast sigmoid/silu helpers (v_rcp_f32 instead of IEEE division ladder)
__device__ __forceinline__ float fast_sigmoid(float x) {
    return __builtin_amdgcn_rcpf(1.f + __expf(-x));
}

// ---------------- GEMM tiling ----------------
#define TM 128
#define TN 128
#define TKK 64
#define LDA_S 72   // padded LDS stride (elements): 144B rows -> 16B aligned, odd-dword spread
#define LDB_S 72

// =====================================================================
// fp32 -> bf16 transpose-convert:  src (R x C fp32, row-major)
//                               -> dst (C x R bf16, row-major)
// 64x64 LDS tiles; both global phases coalesced.
// =====================================================================
#define TT 64
__global__ __launch_bounds__(256) void transpose_cvt_kernel(const float* __restrict__ src,
                                                            __bf16* __restrict__ dst,
                                                            int R, int C)
{
    __shared__ __bf16 tile[TT][TT + 2];
    int r0 = blockIdx.y * TT, c0 = blockIdx.x * TT;
    int t = threadIdx.x;
    #pragma unroll
    for (int q = 0; q < 16; ++q) {
        int idx = t + q * 256;
        int i = idx >> 6, j = idx & 63;
        tile[i][j] = (__bf16)src[(size_t)(r0 + i) * C + c0 + j];
    }
    __syncthreads();
    #pragma unroll
    for (int q = 0; q < 16; ++q) {
        int idx = t + q * 256;
        int i = idx >> 6, j = idx & 63;           // out row c0+i, col r0+j
        dst[(size_t)(c0 + i) * R + r0 + j] = tile[j][i];
    }
}

// =====================================================================
// Router: one wave32 per token: logit = h . w_router ; prob = sigmoid*ALPHA
// =====================================================================
__global__ __launch_bounds__(256) void router_kernel(const float* __restrict__ hidden,
                                                     const float* __restrict__ wr,
                                                     float* __restrict__ logits,
                                                     float* __restrict__ probs)
{
    int tok  = (int)((blockIdx.x * blockDim.x + threadIdx.x) >> 5);
    int lane = threadIdx.x & 31;
    if (tok >= BB * SEQ) return;
    const float* h = hidden + (size_t)tok * DIM;
    float s = 0.f;
    for (int i = lane; i < DIM; i += 32) s += h[i] * wr[i];
    #pragma unroll
    for (int off = 16; off > 0; off >>= 1) s += __shfl_xor(s, off, 32);
    if (lane == 0) {
        logits[tok] = s;
        probs[tok]  = ALPHA * fast_sigmoid(s);
    }
}

// =====================================================================
// z-loss: mean_b( logsumexp(logits_b)^2 )
// =====================================================================
__global__ __launch_bounds__(256) void zloss_kernel(const float* __restrict__ logits,
                                                    float* __restrict__ out_z)
{
    __shared__ float red[256];
    int t = threadIdx.x;
    float acc = 0.f;
    for (int b = 0; b < BB; ++b) {
        const float* l = logits + (size_t)b * SEQ;
        float mx = -INFINITY;
        for (int i = t; i < SEQ; i += 256) mx = fmaxf(mx, l[i]);
        red[t] = mx; __syncthreads();
        for (int off = 128; off > 0; off >>= 1) {
            if (t < off) red[t] = fmaxf(red[t], red[t + off]);
            __syncthreads();
        }
        mx = red[0]; __syncthreads();
        float sm = 0.f;
        for (int i = t; i < SEQ; i += 256) sm += __expf(l[i] - mx);
        red[t] = sm; __syncthreads();
        for (int off = 128; off > 0; off >>= 1) {
            if (t < off) red[t] += red[t + off];
            __syncthreads();
        }
        if (t == 0) { float z = mx + logf(red[0]); acc += z * z; }
        __syncthreads();
    }
    if (t == 0) *out_z = acc / (float)BB;
}

// =====================================================================
// Top-k selection (k = KTOP), tie-break lowest index; sel ascending.
// =====================================================================
__global__ __launch_bounds__(1024) void select_kernel(const float* __restrict__ probs,
                                                      int*   __restrict__ sel,
                                                      float* __restrict__ wts,
                                                      int*   __restrict__ sel_out)
{
    __shared__ float lp[SEQ];
    __shared__ int   part[1024];
    int b = blockIdx.x, t = threadIdx.x;
    const float* p = probs + (size_t)b * SEQ;
    for (int i = t; i < SEQ; i += 1024) lp[i] = p[i];
    __syncthreads();

    int lmask[4]; int lsum = 0;
    #pragma unroll
    for (int q = 0; q < 4; ++q) {
        int i = t * 4 + q;
        float pi = lp[i];
        int r = 0;
        for (int j = 0; j < SEQ; ++j) {
            float pj = lp[j];
            r += (pj > pi) || (pj == pi && j < i);
        }
        lmask[q] = (r < KTOP) ? 1 : 0;
        lsum += lmask[q];
    }
    part[t] = lsum; __syncthreads();
    for (int off = 1; off < 1024; off <<= 1) {
        int v = (t >= off) ? part[t - off] : 0;
        __syncthreads();
        part[t] += v;
        __syncthreads();
    }
    int base = part[t] - lsum;
    #pragma unroll
    for (int q = 0; q < 4; ++q) {
        if (lmask[q]) {
            int i = t * 4 + q;
            size_t o = (size_t)b * KTOP + base;
            sel[o]     = i;
            wts[o]     = lp[i];
            sel_out[o] = i;
            ++base;
        }
    }
}

// =====================================================================
// Gather selected rows -> bf16 A matrix (KTOP x DIM), vectorized
// =====================================================================
__global__ __launch_bounds__(256) void gather_kernel(const float* __restrict__ hidden_b,
                                                     const int* __restrict__ sel_b,
                                                     __bf16* __restrict__ tokA)
{
    int m = blockIdx.x;
    int row = sel_b[m];
    const float4* src = (const float4*)(hidden_b + (size_t)row * DIM);
    bf16x8* dst = (bf16x8*)(tokA + (size_t)m * DIM);
    for (int i = threadIdx.x; i < DIM / 8; i += blockDim.x) {
        float4 a = src[2 * i], b = src[2 * i + 1];
        bf16x8 v = { (__bf16)a.x, (__bf16)a.y, (__bf16)a.z, (__bf16)a.w,
                     (__bf16)b.x, (__bf16)b.y, (__bf16)b.z, (__bf16)b.w };
        dst[i] = v;
    }
}

// =====================================================================
// WMMA mainloop: C_tile(128x128) += A(MxK,row-major) * Bt(NxK,row-major)^T
//   - Bt is the PRE-TRANSPOSED weight (N x K), so both tiles copy as
//     contiguous rows; all LDS traffic is b128.
//   - 8 waves in 4(M) x 2(N); each wave: 32x64 = 2x4 WMMA tiles.
//   - TKK=64 -> 16 WMMAs per barrier pair.
// =====================================================================
__device__ __forceinline__ void gemm_mainloop(const __bf16* __restrict__ A, int lda,
                                              const __bf16* __restrict__ Bt, int ldb,
                                              int K, int m0, int n0,
                                              __bf16* As, __bf16* Bs,
                                              f32x8 acc[2][4])
{
    const int tid  = threadIdx.x;
    const int lane = tid & 31;
    const int wave = tid >> 5;
    const int wm   = (wave & 3) * 32;
    const int wn   = (wave >> 2) * 64;
    const int half = lane >> 4;
    const int l15  = lane & 15;

    const int row  = tid >> 1;            // 0..127
    const int colq = (tid & 1) * 32;      // 0 or 32 (elements)

    for (int k0 = 0; k0 < K; k0 += TKK) {
        // ---- A tile 128x64: 4 x uint4 per thread, contiguous
        {
            const uint4* gsrc = (const uint4*)(A + (size_t)(m0 + row) * lda + k0 + colq);
            uint4* ldst = (uint4*)(As + row * LDA_S + colq);
            #pragma unroll
            for (int i = 0; i < 4; ++i) ldst[i] = gsrc[i];
            __builtin_prefetch(A + (size_t)(m0 + row) * lda + k0 + TKK + colq, 0, 1);
        }
        // ---- B tile 128(n) x 64(k) from pre-transposed weights: contiguous
        {
            const uint4* gsrc = (const uint4*)(Bt + (size_t)(n0 + row) * ldb + k0 + colq);
            uint4* ldst = (uint4*)(Bs + row * LDB_S + colq);
            #pragma unroll
            for (int i = 0; i < 4; ++i) ldst[i] = gsrc[i];
            __builtin_prefetch(Bt + (size_t)(n0 + row) * ldb + k0 + TKK + colq, 0, 1);
        }
        __syncthreads();

        #pragma unroll
        for (int ks = 0; ks < TKK; ks += 32) {
            bf16x16 af[2], bfr[4];
            #pragma unroll
            for (int i = 0; i < 2; ++i) {
                // A frag: lane row M=l15; per-lane K = {half*8..+7, 16+half*8..+7}
                const __bf16* ar = As + (wm + i * 16 + l15) * LDA_S + ks;
                af[i] = cat8(*(const bf16x8*)(ar + half * 8),
                             *(const bf16x8*)(ar + 16 + half * 8));
            }
            #pragma unroll
            for (int j = 0; j < 4; ++j) {
                // B frag: lane col N=l15; per-lane K = {half*8..+7, 16+half*8..+7}
                const __bf16* br = Bs + (wn + j * 16 + l15) * LDB_S + ks + half * 8;
                bfr[j] = cat8(*(const bf16x8*)(br),
                              *(const bf16x8*)(br + 16));
            }
            #pragma unroll
            for (int i = 0; i < 2; ++i)
                #pragma unroll
                for (int j = 0; j < 4; ++j)
                    acc[i][j] = __builtin_amdgcn_wmma_f32_16x16x32_bf16(
                        false, af[i], false, bfr[j], (short)0, acc[i][j], false, false);
        }
        __syncthreads();
    }
}

// =====================================================================
// GEMM1: hmid = bf16( silu( tokA @ w1 ) )    (KTOP x FF)
// =====================================================================
__global__ __launch_bounds__(256) void gemm1_silu_kernel(const __bf16* __restrict__ tokA,
                                                         const __bf16* __restrict__ w1t,
                                                         __bf16* __restrict__ hmid)
{
    __shared__ __bf16 As[TM * LDA_S];
    __shared__ __bf16 Bs[TN * LDB_S];
    int m0 = blockIdx.y * TM, n0 = blockIdx.x * TN;
    f32x8 acc[2][4] = {};
    gemm_mainloop(tokA, DIM, w1t, DIM, DIM, m0, n0, As, Bs, acc);

    int tid = threadIdx.x, lane = tid & 31, wave = tid >> 5;
    int wm = (wave & 3) * 32, wn = (wave >> 2) * 64;
    int half = lane >> 4, l15 = lane & 15;
    #pragma unroll
    for (int i = 0; i < 2; ++i)
        #pragma unroll
        for (int j = 0; j < 4; ++j)
            #pragma unroll
            for (int g = 0; g < 8; ++g) {
                int m = m0 + wm + i * 16 + half * 8 + g;
                int n = n0 + wn + j * 16 + l15;
                float x = acc[i][j][g];
                hmid[(size_t)m * FF + n] = (__bf16)(x * fast_sigmoid(x));
            }
}

// =====================================================================
// GEMM2 + epilogue: delta = hmid @ w2 ; out[sel[m]] += (h[sel[m]] + delta)*wt[m]
// =====================================================================
__global__ __launch_bounds__(256) void gemm2_scatter_kernel(const __bf16* __restrict__ hmid,
                                                            const __bf16* __restrict__ w2t,
                                                            const float* __restrict__ hidden_b,
                                                            float* __restrict__ out_b,
                                                            const int*  __restrict__ sel_b,
                                                            const float* __restrict__ wts_b)
{
    __shared__ __bf16 As[TM * LDA_S];
    __shared__ __bf16 Bs[TN * LDB_S];
    int m0 = blockIdx.y * TM, n0 = blockIdx.x * TN;
    f32x8 acc[2][4] = {};
    gemm_mainloop(hmid, FF, w2t, FF, FF, m0, n0, As, Bs, acc);

    int tid = threadIdx.x, lane = tid & 31, wave = tid >> 5;
    int wm = (wave & 3) * 32, wn = (wave >> 2) * 64;
    int half = lane >> 4, l15 = lane & 15;
    #pragma unroll
    for (int i = 0; i < 2; ++i)
        #pragma unroll
        for (int j = 0; j < 4; ++j)
            #pragma unroll
            for (int g = 0; g < 8; ++g) {
                int m = m0 + wm + i * 16 + half * 8 + g;
                int n = n0 + wn + j * 16 + l15;
                int row = sel_b[m];
                float w = wts_b[m];
                size_t off = (size_t)row * DIM + n;
                float h = hidden_b[off];
                out_b[off] += (h + acc[i][j][g]) * w;   // unique (row,n): race-free
            }
}

// =====================================================================
// host-side launch
// =====================================================================
extern "C" void kernel_launch(void* const* d_in, const int* in_sizes, int n_in,
                              void* d_out, int out_size, void* d_ws, size_t ws_size,
                              hipStream_t stream)
{
    (void)in_sizes; (void)n_in; (void)out_size; (void)ws_size;
    const float* hidden = (const float*)d_in[0];
    const float* wr     = (const float*)d_in[1];
    const float* w1     = (const float*)d_in[2];
    const float* w2     = (const float*)d_in[3];
    float* out = (float*)d_out;

    char* ws = (char*)d_ws;
    size_t off = 0;
    auto carve = [&](size_t bytes) -> char* {
        char* p = ws + off;
        off = (off + bytes + 255) & ~(size_t)255;
        return p;
    };
    float*  logits = (float*) carve((size_t)BB * SEQ * sizeof(float));
    float*  probs  = (float*) carve((size_t)BB * SEQ * sizeof(float));
    int*    sel    = (int*)   carve((size_t)BB * KTOP * sizeof(int));
    float*  wts    = (float*) carve((size_t)BB * KTOP * sizeof(float));
    __bf16* w1t    = (__bf16*)carve((size_t)DIM * FF * sizeof(__bf16));  // FF x DIM
    __bf16* w2t    = (__bf16*)carve((size_t)FF * DIM * sizeof(__bf16));  // DIM x FF
    __bf16* tokA   = (__bf16*)carve((size_t)KTOP * DIM * sizeof(__bf16));
    __bf16* hmid   = (__bf16*)carve((size_t)KTOP * FF * sizeof(__bf16));

    // 1) one-time transpose + bf16 downconversion of the weights
    {
        dim3 g1(FF / TT, DIM / TT);   // w1 (DIM x FF) -> w1t (FF x DIM)
        transpose_cvt_kernel<<<g1, 256, 0, stream>>>(w1, w1t, DIM, FF);
        dim3 g2(DIM / TT, FF / TT);   // w2 (FF x DIM) -> w2t (DIM x FF)
        transpose_cvt_kernel<<<g2, 256, 0, stream>>>(w2, w2t, FF, DIM);
    }

    // 2) router logits/probs
    router_kernel<<<(BB * SEQ * 32) / 256, 256, 0, stream>>>(hidden, wr, logits, probs);

    // 3) z-loss scalar -> d_out[B*S*D]
    zloss_kernel<<<1, 256, 0, stream>>>(logits, out + (size_t)BB * SEQ * DIM);

    // 4) top-k selection; sel also written (int bits) to d_out tail
    int* sel_out = (int*)(out + (size_t)BB * SEQ * DIM + 1);
    select_kernel<<<BB, 1024, 0, stream>>>(probs, sel, wts, sel_out);

    // 5) out = hidden (scatter-add target)
    hipMemcpyAsync(out, hidden, (size_t)BB * SEQ * DIM * sizeof(float),
                   hipMemcpyDeviceToDevice, stream);

    // 6) per-batch FFN
    for (int b = 0; b < BB; ++b) {
        const float* hb   = hidden + (size_t)b * SEQ * DIM;
        float*       ob   = out    + (size_t)b * SEQ * DIM;
        const int*   selb = sel    + (size_t)b * KTOP;
        const float* wtsb = wts    + (size_t)b * KTOP;

        gather_kernel<<<KTOP, 256, 0, stream>>>(hb, selb, tokA);

        dim3 g1(FF / TN, KTOP / TM);    // 64 x 16
        gemm1_silu_kernel<<<g1, 256, 0, stream>>>(tokA, w1t, hmid);

        dim3 g2(DIM / TN, KTOP / TM);   // 16 x 16
        gemm2_scatter_kernel<<<g2, 256, 0, stream>>>(hmid, w2t, hb, ob, selb, wtsb);
    }
}
